// GRU_52355651338739
// MI455X (gfx1250) — matrix operand order repaired and checked by reading, not compiled, not run on hardware
//
#include <hip/hip_runtime.h>
#include <math.h>

// ---------------------------------------------------------------------------
// 2-layer GRU for MI455X (gfx1250, wave32, WMMA).
//  * bf16 WMMA (v_wmma_f32_16x16x32_bf16) for all GEMMs, f32 accumulate.
//  * Input-side gate GEMMs hoisted over T (parallel, compute-bound).
//  * Recurrent part: persistent kernel, 32 WGs each owning 16 hidden units
//    (its r/z/n gate columns), per-timestep device-wide barrier via atomics.
//    - W_hh slice (48KB) staged into LDS once; hot loop reads B-frags via ds.
//    - f32 hidden state kept entirely in registers (matches C-frag layout).
//    - bf16 h ping-pong buffers (32KB) in global for cross-WG A-fragments.
//    - global_prefetch of next step's xg rows overlaps the barrier wait.
// ---------------------------------------------------------------------------

typedef __bf16 bf16_t;
typedef __attribute__((ext_vector_type(16))) __bf16 v16bf;
typedef __attribute__((ext_vector_type(8)))  __bf16 v8bf;
typedef __attribute__((ext_vector_type(8)))  float  v8f;

#define GB 32      // batch
#define GT 2048    // time steps
#define GD 256     // input dim
#define GH 512     // hidden dim
#define GG 1536    // 3*H (gates r,z,n)

// ---- helpers --------------------------------------------------------------

__device__ __forceinline__ bf16_t f2bf(float f) {
  union { float f; unsigned u; } x; x.f = f;
  unsigned r = (x.u + 0x7FFFu + ((x.u >> 16) & 1u)) >> 16;  // RNE
  union { unsigned short s; bf16_t b; } y; y.s = (unsigned short)r;
  return y.b;
}

// A-matrix 16x32 bf16 fragment: lane<16 holds row m0+lane,
// K = [k0..k0+16); lane>=16 holds K = [k0+16..k0+32) of row m0+(lane-16).
__device__ __forceinline__ v16bf load_a_frag(const bf16_t* __restrict__ base,
                                             size_t row_stride, int m0, int k0) {
  const int lane = threadIdx.x & 31;
  const bf16_t* rp = base + (size_t)(m0 + (lane & 15)) * row_stride
                          + (size_t)(k0 + ((lane & 16) ? 16 : 0));
  v8bf lo = *(const v8bf*)(rp);
  v8bf hi = *(const v8bf*)(rp + 8);
  return __builtin_shufflevector(lo, hi, 0,1,2,3,4,5,6,7,8,9,10,11,12,13,14,15);
}

// B-matrix 32x16 bf16 fragment: lane<16 = column n0+lane, K=[k0..k0+16);
// lane>=16 = same columns, K=[k0+16..k0+32). Row-major (n,k) source -> one
// contiguous 32B read per lane.
__device__ __forceinline__ v16bf load_b_frag(const bf16_t* __restrict__ base,
                                             size_t row_stride, int n0, int k0) {
  const int lane = threadIdx.x & 31;
  const bf16_t* rp = base + (size_t)(n0 + (lane & 15)) * row_stride
                          + (size_t)(k0 + ((lane & 16) ? 16 : 0));
  return *(const v16bf*)rp;
}

// Same fragment pattern, LDS source (row stride GH, local n base 0).
__device__ __forceinline__ v16bf load_b_frag_lds(const bf16_t* base, int k0) {
  const int lane = threadIdx.x & 31;
  const bf16_t* rp = base + (size_t)(lane & 15) * GH
                          + (size_t)(k0 + ((lane & 16) ? 16 : 0));
  return *(const v16bf*)rp;
}

__device__ __forceinline__ v8f wmma_bf16(v16bf a, v16bf b, v8f c) {
  return __builtin_amdgcn_wmma_f32_16x16x32_bf16(false, a, false, b,
                                                 (short)0, c, false, false);
}

// Device-wide barrier: monotonically increasing counter, target = nwg*epoch.
__device__ __forceinline__ void grid_barrier(unsigned* cnt, unsigned target) {
  __syncthreads();
  __threadfence();                       // release our h writes (device scope)
  if (threadIdx.x == 0) {
    atomicAdd(cnt, 1u);
    while (atomicAdd(cnt, 0u) < target) __builtin_amdgcn_s_sleep(1);
  }
  __syncthreads();
  __threadfence();                       // acquire remote h writes
}

// ---- elementwise kernels --------------------------------------------------

__global__ void convert_f32_bf16(const float* __restrict__ src,
                                 bf16_t* __restrict__ dst, int n) {
  int i = blockIdx.x * blockDim.x + threadIdx.x;
  int stride = gridDim.x * blockDim.x;
  for (; i < n; i += stride) dst[i] = f2bf(src[i]);
}

__global__ void zero_f32(float* __restrict__ p, int n) {
  int i = blockIdx.x * blockDim.x + threadIdx.x;
  int stride = gridDim.x * blockDim.x;
  for (; i < n; i += stride) p[i] = 0.f;
}

// ---- batched gate GEMM: xg[t] = A[t] @ W^T + bias, out (T,B,3H) f32 -------
// Block: 256 threads = 8 waves; each wave owns 32 gate columns (2 N tiles),
// both 16-row M tiles (B=32). grid = (GG/256, T).

__global__ __launch_bounds__(256)
void gemm_gates(const bf16_t* __restrict__ A, size_t a_b_stride, size_t a_t_stride,
                int K, const bf16_t* __restrict__ W, const float* __restrict__ bias,
                float* __restrict__ out_xg) {
  const int t = blockIdx.y;
  const int wave = threadIdx.x >> 5;
  const int n_base = blockIdx.x * 256 + wave * 32;
  const bf16_t* At = A + (size_t)t * a_t_stride;

  v8f c00 = {}, c01 = {}, c10 = {}, c11 = {};
  for (int k0 = 0; k0 < K; k0 += 32) {
    v16bf a0 = load_a_frag(At, a_b_stride, 0,  k0);
    v16bf a1 = load_a_frag(At, a_b_stride, 16, k0);
    v16bf b0 = load_b_frag(W, (size_t)K, n_base,      k0);
    v16bf b1 = load_b_frag(W, (size_t)K, n_base + 16, k0);
    c00 = wmma_bf16(a0, b0, c00);
    c10 = wmma_bf16(a1, b0, c10);
    c01 = wmma_bf16(a0, b1, c01);
    c11 = wmma_bf16(a1, b1, c11);
  }

  const int lane = threadIdx.x & 31;
  const int n_in = lane & 15;
  const int mh   = (lane >> 4) * 8;      // C layout: lanes16-31 hold M+8
  float* orow = out_xg + (size_t)t * GB * GG;
  #pragma unroll
  for (int nt = 0; nt < 2; ++nt) {
    int n = n_base + nt * 16 + n_in;
    float bv = bias[n];
    v8f cm0 = nt ? c01 : c00;
    v8f cm1 = nt ? c11 : c10;
    #pragma unroll
    for (int v = 0; v < 8; ++v) {
      orow[(size_t)(mh + v) * GG + n]      = cm0[v] + bv;
      orow[(size_t)(16 + mh + v) * GG + n] = cm1[v] + bv;
    }
  }
}

// ---- persistent recurrent kernel ------------------------------------------
// 32 WGs x 64 threads (2 waves). WG owns hidden units j in [16*bx, 16*bx+16):
// gate columns j, H+j, 2H+j -> r/z/n combine is WG-local, one barrier/step.
// Wave 0 computes batch rows 0..15, wave 1 rows 16..31.

__global__ __launch_bounds__(64)
void gru_recurrent(const float* __restrict__ xg,        // (T,B,3H) f32
                   const bf16_t* __restrict__ Whh,      // (3H,H) bf16
                   const float* __restrict__ bhh,       // (3H) f32
                   bf16_t* __restrict__ hA, bf16_t* __restrict__ hB, // ping-pong
                   bf16_t* __restrict__ h1seq,          // (T,B,H) bf16 or null
                   float* __restrict__ sumbuf,          // (T,B) f32 or null
                   unsigned* __restrict__ bar, int nwg) {
  const int j0   = blockIdx.x * 16;
  const int wave = threadIdx.x >> 5;
  const int m0   = wave * 16;
  const int lane = threadIdx.x & 31;

  // --- stage this WG's W_hh slice into LDS: 3 gates x 16 rows x 512 k -----
  // Rows [j0, j0+16) of each gate block are contiguous (row stride = GH).
  __shared__ bf16_t sW[3 * 16 * GH];                       // 48 KB
  #pragma unroll
  for (int g = 0; g < 3; ++g) {
    const bf16_t* src = Whh + ((size_t)g * GH + j0) * GH;  // 16 KB contiguous
    bf16_t* dst = sW + g * 16 * GH;
    for (int e = threadIdx.x * 8; e < 16 * GH; e += 64 * 8)
      *(v8bf*)(dst + e) = *(const v8bf*)(src + e);
  }

  // zero this WG's slice of the bf16 h ping-pong buffers
  for (int idx = threadIdx.x; idx < GB * 16; idx += 64) {
    int b = idx >> 4;
    int j = j0 + (idx & 15);
    hA[(size_t)b * GH + j] = f2bf(0.f);
    hB[(size_t)b * GH + j] = f2bf(0.f);
  }
  unsigned epoch = 1;
  grid_barrier(bar, (unsigned)nwg * epoch);    // includes __syncthreads for sW

  const int n_in  = lane & 15;
  const int j     = j0 + n_in;
  const int mbase = m0 + (lane >> 4) * 8;
  const float bhr = bhh[j], bhz = bhh[GH + j], bhn = bhh[2 * GH + j];

  // f32 hidden state lives in registers: hreg[v] == h[mbase+v][j]
  v8f hreg = {};

  for (int t = 0; t < GT; ++t) {
    const bf16_t* hin  = (t & 1) ? hB : hA;
    bf16_t*       hout = (t & 1) ? hA : hB;

    v8f cr = {}, cz = {}, cn = {};
    #pragma unroll 4
    for (int k0 = 0; k0 < GH; k0 += 32) {
      v16bf a  = load_a_frag(hin, GH, m0, k0);
      v16bf br = load_b_frag_lds(sW,               k0);
      v16bf bz = load_b_frag_lds(sW + 16 * GH,     k0);
      v16bf bn = load_b_frag_lds(sW + 2 * 16 * GH, k0);
      cr = wmma_bf16(a, br, cr);
      cz = wmma_bf16(a, bz, cz);
      cn = wmma_bf16(a, bn, cn);
    }

    const float* xgt = xg + (size_t)t * GB * GG;
    #pragma unroll
    for (int v = 0; v < 8; ++v) {
      int b = mbase + v;
      const float* xrow = xgt + (size_t)b * GG;
      float r  = 1.f / (1.f + expf(-(xrow[j]          + cr[v] + bhr)));
      float z  = 1.f / (1.f + expf(-(xrow[GH + j]     + cz[v] + bhz)));
      float nn = tanhf(        xrow[2 * GH + j] + r * (cn[v] + bhn));
      float hnew = (1.f - z) * nn + z * hreg[v];
      hreg[v] = hnew;
      hout[(size_t)b * GH + j] = f2bf(hnew);
      if (h1seq) h1seq[((size_t)t * GB + b) * GH + j] = f2bf(hnew);
      if (sumbuf) {  // reduce over the 16 j-lanes sharing this (t,b)
        float s = hnew;
        s += __shfl_xor(s, 1, 16);
        s += __shfl_xor(s, 2, 16);
        s += __shfl_xor(s, 4, 16);
        s += __shfl_xor(s, 8, 16);
        if (n_in == 0) atomicAdd(&sumbuf[(size_t)t * GB + b], s);
      }
    }

    // prefetch next step's xg rows (read-only) so the fetch overlaps the
    // barrier wait; emits global_prefetch_b8.
    if (t + 1 < GT) {
      const float* nxt = xg + (size_t)(t + 1) * GB * GG;
      #pragma unroll
      for (int v = 0; v < 8; ++v) {
        const float* xrow = nxt + (size_t)(mbase + v) * GG;
        __builtin_prefetch(xrow + j,          0, 1);
        __builtin_prefetch(xrow + GH + j,     0, 1);
        __builtin_prefetch(xrow + 2 * GH + j, 0, 1);
      }
    }

    epoch++;
    grid_barrier(bar, (unsigned)nwg * epoch);
  }
}

// ---- finalize: out[b,t] = |sum_h(h2) * k| ---------------------------------

__global__ void finalize_out(const float* __restrict__ sumbuf,
                             const float* __restrict__ kptr,
                             float* __restrict__ out) {
  int i = blockIdx.x * blockDim.x + threadIdx.x;  // over B*T
  if (i >= GB * GT) return;
  int b = i / GT, t = i % GT;
  out[i] = fabsf(sumbuf[(size_t)t * GB + b] * kptr[0]);
}

// ---- launch ---------------------------------------------------------------

extern "C" void kernel_launch(void* const* d_in, const int* in_sizes, int n_in_cnt,
                              void* d_out, int out_size, void* d_ws, size_t ws_size,
                              hipStream_t stream) {
  (void)in_sizes; (void)n_in_cnt; (void)out_size; (void)ws_size;
  const float* x    = (const float*)d_in[0];
  const float* Wih0 = (const float*)d_in[1];
  const float* Whh0 = (const float*)d_in[2];
  const float* bih0 = (const float*)d_in[3];
  const float* bhh0 = (const float*)d_in[4];
  const float* Wih1 = (const float*)d_in[5];
  const float* Whh1 = (const float*)d_in[6];
  const float* bih1 = (const float*)d_in[7];
  const float* bhh1 = (const float*)d_in[8];
  const float* kptr = (const float*)d_in[9];
  float* out = (float*)d_out;

  // workspace carve (256B aligned)
  char* p = (char*)d_ws;
  auto carve = [&](size_t bytes) { char* r = p; p += (bytes + 255) & ~(size_t)255; return r; };
  float*    xg      = (float*)  carve((size_t)GT * GB * GG * 4);   // 403 MB
  bf16_t*   x_bf    = (bf16_t*) carve((size_t)GB * GT * GD * 2);   // 32 MB
  bf16_t*   h1_bf   = (bf16_t*) carve((size_t)GT * GB * GH * 2);   // 64 MB
  bf16_t*   Wih0_bf = (bf16_t*) carve((size_t)GG * GD * 2);
  bf16_t*   Whh0_bf = (bf16_t*) carve((size_t)GG * GH * 2);
  bf16_t*   Wih1_bf = (bf16_t*) carve((size_t)GG * GH * 2);
  bf16_t*   Whh1_bf = (bf16_t*) carve((size_t)GG * GH * 2);
  bf16_t*   hA      = (bf16_t*) carve((size_t)GB * GH * 2);
  bf16_t*   hB      = (bf16_t*) carve((size_t)GB * GH * 2);
  float*    sumbuf  = (float*)  carve((size_t)GT * GB * 4);
  unsigned* bars    = (unsigned*)carve(256);

  // 1) precision convert (f32 -> bf16) for WMMA operands
  convert_f32_bf16<<<2048, 256, 0, stream>>>(x,    x_bf,    GB * GT * GD);
  convert_f32_bf16<<<512,  256, 0, stream>>>(Wih0, Wih0_bf, GG * GD);
  convert_f32_bf16<<<512,  256, 0, stream>>>(Whh0, Whh0_bf, GG * GH);
  convert_f32_bf16<<<512,  256, 0, stream>>>(Wih1, Wih1_bf, GG * GH);
  convert_f32_bf16<<<512,  256, 0, stream>>>(Whh1, Whh1_bf, GG * GH);
  // 2) zero accumulators / barrier counters (fresh every call -> deterministic)
  zero_f32<<<256, 256, 0, stream>>>(sumbuf, GT * GB);
  zero_f32<<<1,   64,  0, stream>>>((float*)bars, 64);

  dim3 gemm_grid(GG / 256, GT);
  const int NWG = GH / 16;  // 32 persistent workgroups

  // 3) layer0 input gates: xg0 = x @ Wih0^T + bih0   (A row = x[b,t,:], K=D)
  gemm_gates<<<gemm_grid, 256, 0, stream>>>(x_bf, (size_t)GT * GD, (size_t)GD,
                                            GD, Wih0_bf, bih0, xg);
  // 4) layer0 recurrence -> h1 sequence (bf16)
  gru_recurrent<<<NWG, 64, 0, stream>>>(xg, Whh0_bf, bhh0, hA, hB,
                                        h1_bf, nullptr, bars + 0, NWG);
  // 5) layer1 input gates: xg1 = h1 @ Wih1^T + bih1  (A row = h1[t,b,:], K=H)
  gemm_gates<<<gemm_grid, 256, 0, stream>>>(h1_bf, (size_t)GH, (size_t)GB * GH,
                                            GH, Wih1_bf, bih1, xg);
  // 6) layer1 recurrence -> per-(t,b) hidden sums
  gru_recurrent<<<NWG, 64, 0, stream>>>(xg, Whh1_bf, bhh1, hA, hB,
                                        nullptr, sumbuf, bars + 32, NWG);
  // 7) out[b,t] = |sum * k|
  finalize_out<<<(GB * GT + 255) / 256, 256, 0, stream>>>(sumbuf, kptr, out);
}
